// BaselineModel_21199958573367
// MI455X (gfx1250) — matrix operand order repaired and checked by reading, not compiled, not run on hardware
//
#include <hip/hip_runtime.h>
#include <hip/hip_bf16.h>

typedef __attribute__((ext_vector_type(16))) _Float16 v16h;
typedef __attribute__((ext_vector_type(8)))  _Float16 v8h;
typedef __attribute__((ext_vector_type(8)))  float    v8f;

// LDS byte-offset of a pointer into a __shared__ array (addrspace(3) cast).
#define LDS_OFF(p) ((unsigned)(size_t)(__attribute__((address_space(3))) const void*)(p))

// ---------------------------------------------------------------------------
// Utility kernels
// ---------------------------------------------------------------------------

__global__ __launch_bounds__(256) void zero_f32_kernel(float* p, int n) {
    int i = blockIdx.x * 256 + threadIdx.x;
    if (i < n) p[i] = 0.0f;
}

// Repack conv weights: OIHW f32 -> [tap][O][I] f16  (tap = ky*5+kx)
__global__ __launch_bounds__(256) void pack_w_kernel(const float* __restrict__ w,
                                                     _Float16* __restrict__ wp,
                                                     int O, int I) {
    int i = blockIdx.x * 256 + threadIdx.x;
    int total = O * I * 25;
    if (i >= total) return;
    int ci = i % I;
    int t  = i / I;
    int o  = t % O;
    int tap = t / O;
    wp[i] = (_Float16)w[(size_t)(o * I + ci) * 25 + tap];
}

// 2x2 max-pool on channels-last f16 [B][H][W][C] -> [B][H/2][W/2][C]
__global__ __launch_bounds__(256) void pool2_kernel(const _Float16* __restrict__ in,
                                                    _Float16* __restrict__ out,
                                                    int H, int W, int C, int total) {
    int i = blockIdx.x * 256 + threadIdx.x;
    if (i >= total) return;
    int c = i % C;
    int t = i / C;
    int xo = t % (W / 2); t /= (W / 2);
    int yo = t % (H / 2);
    int b  = t / (H / 2);
    const _Float16* p0 = in + ((size_t)(b * H + yo * 2) * W + xo * 2) * C + c;
    const _Float16* p1 = p0 + (size_t)W * C;
    float m = fmaxf(fmaxf((float)p0[0], (float)p0[C]),
                    fmaxf((float)p1[0], (float)p1[C]));
    out[i] = (_Float16)m;
}

// ---------------------------------------------------------------------------
// Conv1: Cin=1, direct conv fused bias+ReLU+2x2pool.
// x: [16][1][220][220] f32 -> act1: [16][108][108][64] f16 (channels-last)
// ---------------------------------------------------------------------------
__global__ __launch_bounds__(256) void conv1_kernel(const float* __restrict__ x,
                                                    const float* __restrict__ w1,
                                                    const float* __restrict__ b1,
                                                    _Float16* __restrict__ out) {
    __shared__ float ws[64 * 25];
    __shared__ float bs[64];
    int tid = threadIdx.x;
    for (int i = tid; i < 64 * 25; i += 256) ws[i] = w1[i];  // OIHW == [c][25]
    if (tid < 64) bs[tid] = b1[tid];
    __syncthreads();

    int gid = blockIdx.x * 256 + tid;
    if (gid >= 16 * 108 * 108) return;
    int px = gid % 108;
    int t  = gid / 108;
    int py = t % 108;
    int b  = t / 108;

    const float* xp = x + ((size_t)b * 220 + py * 2) * 220 + px * 2;
    float patch[36];
#pragma unroll
    for (int yy = 0; yy < 6; ++yy)
#pragma unroll
        for (int xx = 0; xx < 6; ++xx)
            patch[yy * 6 + xx] = xp[yy * 220 + xx];

    _Float16* dst = out + (size_t)gid * 64;
    for (int c = 0; c < 64; ++c) {
        float a00 = bs[c], a01 = bs[c], a10 = bs[c], a11 = bs[c];
#pragma unroll
        for (int ky = 0; ky < 5; ++ky)
#pragma unroll
            for (int kx = 0; kx < 5; ++kx) {
                float wv = ws[c * 25 + ky * 5 + kx];
                a00 = fmaf(wv, patch[ky * 6 + kx],       a00);
                a01 = fmaf(wv, patch[ky * 6 + kx + 1],   a01);
                a10 = fmaf(wv, patch[(ky + 1) * 6 + kx], a10);
                a11 = fmaf(wv, patch[(ky + 1) * 6 + kx + 1], a11);
            }
        float m = fmaxf(fmaxf(a00, a01), fmaxf(a10, a11));
        m = fmaxf(m, 0.0f);  // relu(max) == max(relu)
        dst[c] = (_Float16)m;
    }
}

// ---------------------------------------------------------------------------
// Implicit-GEMM conv with WMMA f16->f32 (16x16x32), async global->LDS staging.
// act: [B][HIN][WIN][CIN] f16 (channels-last), wpack: [25][COUT][CIN] f16
// out: [B][HOUT][WOUT][COUT] f16 = relu(conv + bias)
// Workgroup: 256 thr (8 waves) -> 128 couts x 64 output positions.
// Double-buffered LDS: next K-slice streamed via GLOBAL_LOAD_ASYNC_TO_LDS_B128
// while WMMAs consume the current buffer; one barrier per K-step.
// ---------------------------------------------------------------------------
template <int CIN, int COUT, int HIN, int WIN>
__global__ __launch_bounds__(256) void conv_wmma_kernel(const _Float16* __restrict__ act,
                                                        const _Float16* __restrict__ wpack,
                                                        const float* __restrict__ bias,
                                                        _Float16* __restrict__ out) {
    constexpr int HOUT  = HIN - 4;
    constexpr int WOUT  = WIN - 4;
    constexpr int KH    = CIN / 32;   // 32-wide K slices per tap
    constexpr int STEPS = 25 * KH;

    __shared__ __align__(32) _Float16 ldsA[2][128 * 32];  // [m][k] row-major
    __shared__ __align__(32) _Float16 ldsB[2][64 * 32];   // [n][k] row-major

    const int tid  = threadIdx.x;
    const int wave = tid >> 5;
    const int lane = tid & 31;
    const int lrow = lane & 15;   // row/col within 16
    const int lhi  = lane >> 4;   // half-wave select
    const int b     = blockIdx.z;
    const int mblk  = blockIdx.y * 128;
    const int ntile = blockIdx.x * 64;

    v8f acc[4];
#pragma unroll
    for (int i = 0; i < 4; ++i) acc[i] = (v8f)(0.0f);

    // Issue async fills for K-step `step` into buffer `bufsel`.
    auto stage = [&](int step, int bufsel) {
        int tap = step / KH;
        int h   = step - tap * KH;
        int dy = tap / 5, dx = tap % 5;
        // A tile: 128 x 32 halves = 512 x 16B; 2 async b128 per thread
#pragma unroll
        for (int r = 0; r < 2; ++r) {
            int idx   = tid + r * 256;
            int row   = idx >> 2;
            int chunk = idx & 3;
            const _Float16* src = wpack +
                ((size_t)(tap * COUT + mblk + row)) * CIN + h * 32 + chunk * 8;
            unsigned dst = LDS_OFF(&ldsA[bufsel][row * 32 + chunk * 8]);
            asm volatile("global_load_async_to_lds_b128 %0, %1, off"
                         :: "v"(dst), "v"(src) : "memory");
        }
        // B tile (im2col): 64 positions x 32 cin; 1 async b128 per thread
        {
            int n     = tid >> 2;
            int chunk = tid & 3;
            int pos = ntile + n;
            int oy  = pos / WOUT;
            int ox  = pos - oy * WOUT;
            const _Float16* src = act +
                ((size_t)(b * HIN + oy + dy) * WIN + (ox + dx)) * CIN + h * 32 + chunk * 8;
            unsigned dst = LDS_OFF(&ldsB[bufsel][n * 32 + chunk * 8]);
            asm volatile("global_load_async_to_lds_b128 %0, %1, off"
                         :: "v"(dst), "v"(src) : "memory");
        }
    };

    stage(0, 0);

#pragma unroll 1
    for (int step = 0; step < STEPS; ++step) {
        const int buf = step & 1;
        // own async fills done -> barrier makes every wave's fills visible
        asm volatile("s_wait_asynccnt 0x0" ::: "memory");
        __syncthreads();
        if (step + 1 < STEPS) stage(step + 1, buf ^ 1);

        // A fragment: lane row m = msub + lrow; K contiguous per half-wave (matches B)
        v16h afrag;
        {
            int m  = (wave << 4) + lrow;
            afrag = *(const v16h*)(&ldsA[buf][m * 32 + lhi * 16]);  // K = 16*lhi..16*lhi+15
        }
        // 4 B fragments / WMMAs (B: col = lane&15, K contiguous per half-wave)
#pragma unroll
        for (int ns = 0; ns < 4; ++ns) {
            v16h bfrag = *(const v16h*)(&ldsB[buf][(ns * 16 + lrow) * 32 + lhi * 16]);
            acc[ns] = __builtin_amdgcn_wmma_f32_16x16x32_f16(
                false, afrag, false, bfrag, (short)0, acc[ns], false, false);
        }
    }

    // --- epilogue: bias + relu, f16 store (channels-last)
#pragma unroll
    for (int ns = 0; ns < 4; ++ns) {
        int pos = ntile + ns * 16 + lrow;
        int oy  = pos / WOUT;
        int ox  = pos - oy * WOUT;
        int cbase = mblk + (wave << 4) + lhi * 8;
        _Float16* dst = out + ((size_t)(b * HOUT + oy) * WOUT + ox) * COUT + cbase;
        v8h o;
#pragma unroll
        for (int r = 0; r < 8; ++r) {
            float v = acc[ns][r] + bias[cbase + r];
            o[r] = (_Float16)fmaxf(v, 0.0f);
        }
        *(v8h*)dst = o;
    }
}

// ---------------------------------------------------------------------------
// FC: g[b][cout] = sum_k flat(b,k) * fcw[cout][k]   (K = 147456, split x72)
// flat(b,k) with k = c*576 + r maps to act3[b*147456 + r*256 + c] (chan-last).
// HBM-bound on fcw (302 MB) -> coalesced f32 reads + prefetch, K-split +
// atomicAdd into g.
// ---------------------------------------------------------------------------
__global__ __launch_bounds__(256) void fc_kernel(const float* __restrict__ fcw,
                                                 const _Float16* __restrict__ act3,
                                                 float* __restrict__ g) {
    const int cout = blockIdx.x;
    const int k0   = blockIdx.y * 2048;
    const int tid  = threadIdx.x;

    float acc[16];
#pragma unroll
    for (int b = 0; b < 16; ++b) acc[b] = 0.0f;

    const float* wrow = fcw + (size_t)cout * 147456 + k0;
#pragma unroll 1
    for (int j = 0; j < 8; ++j) {
        int k = tid + j * 256;
        if (j + 1 < 8) __builtin_prefetch(&wrow[k + 256], 0, 0);  // global_prefetch_b8
        float wv = wrow[k];
        int kk = k0 + k;
        int c  = kk / 576;
        int r  = kk - c * 576;
        const _Float16* ap = act3 + r * 256 + c;
#pragma unroll
        for (int b = 0; b < 16; ++b)
            acc[b] = fmaf(wv, (float)ap[(size_t)b * 147456], acc[b]);
    }

    __shared__ float red[256];
    for (int b = 0; b < 16; ++b) {
        red[tid] = acc[b];
        __syncthreads();
        for (int s = 128; s > 0; s >>= 1) {
            if (tid < s) red[tid] += red[tid + s];
            __syncthreads();
        }
        if (tid == 0) atomicAdd(&g[b * 512 + cout], red[0]);
        __syncthreads();
    }
}

// ---------------------------------------------------------------------------
// Attention: one block per batch. feat channels-last f16: f(b,c,n)=feat[(b*HW+n)*C+c]
// q = aw @ (g+fcb) + ab ; softmax over n in LDS ; out[c] = sum_n f*p
// ---------------------------------------------------------------------------
__global__ __launch_bounds__(256) void att_kernel(const _Float16* __restrict__ feat,
                                                  int C, int HW,
                                                  const float* __restrict__ aw,
                                                  const float* __restrict__ ab,
                                                  const float* __restrict__ g_raw,
                                                  const float* __restrict__ fcb,
                                                  float* __restrict__ outp,
                                                  int outOff) {
    __shared__ float gl[512];
    __shared__ float q[256];
    __shared__ float sc[11664];     // max spatial N (stage 1)
    __shared__ float red[256];

    const int tid = threadIdx.x;
    const int b   = blockIdx.x;

    gl[tid]       = g_raw[b * 512 + tid]       + fcb[tid];
    gl[tid + 256] = g_raw[b * 512 + tid + 256] + fcb[tid + 256];
    __syncthreads();

    if (tid < C) {
        float s = ab[tid];
        const float* ar = aw + (size_t)tid * 512;
        for (int j = 0; j < 512; ++j) s = fmaf(ar[j], gl[j], s);
        q[tid] = s;
    }
    __syncthreads();

    const _Float16* fb = feat + (size_t)b * HW * C;
    float lmax = -3.4e38f;
    for (int n = tid; n < HW; n += 256) {
        const _Float16* fp = fb + (size_t)n * C;
        float s = 0.0f;
        for (int c = 0; c < C; ++c) s = fmaf((float)fp[c], q[c], s);
        sc[n] = s;
        lmax = fmaxf(lmax, s);
    }
    red[tid] = lmax;
    __syncthreads();
    for (int s = 128; s > 0; s >>= 1) {
        if (tid < s) red[tid] = fmaxf(red[tid], red[tid + s]);
        __syncthreads();
    }
    float mx = red[0];
    __syncthreads();

    float lsum = 0.0f;
    for (int n = tid; n < HW; n += 256) {
        float e = __expf(sc[n] - mx);
        sc[n] = e;
        lsum += e;
    }
    red[tid] = lsum;
    __syncthreads();
    for (int s = 128; s > 0; s >>= 1) {
        if (tid < s) red[tid] += red[tid + s];
        __syncthreads();
    }
    float inv = 1.0f / red[0];
    __syncthreads();

    if (tid < C) {
        float a = 0.0f;
        for (int n = 0; n < HW; ++n)
            a = fmaf((float)fb[(size_t)n * C + tid], sc[n], a);   // lane-coalesced in c
        outp[b * 448 + outOff + tid] = a * inv;
    }
}

// ---------------------------------------------------------------------------
// Launch
// ---------------------------------------------------------------------------
extern "C" void kernel_launch(void* const* d_in, const int* in_sizes, int n_in,
                              void* d_out, int out_size, void* d_ws, size_t ws_size,
                              hipStream_t stream) {
    (void)in_sizes; (void)n_in; (void)out_size; (void)ws_size;

    const float* x   = (const float*)d_in[0];
    const float* w1  = (const float*)d_in[1];
    const float* b1  = (const float*)d_in[2];
    const float* w2  = (const float*)d_in[3];
    const float* b2  = (const float*)d_in[4];
    const float* w3  = (const float*)d_in[5];
    const float* b3  = (const float*)d_in[6];
    const float* fcw = (const float*)d_in[7];
    const float* fcb = (const float*)d_in[8];
    const float* aw1 = (const float*)d_in[9];
    const float* ab1 = (const float*)d_in[10];
    const float* aw2 = (const float*)d_in[11];
    const float* ab2 = (const float*)d_in[12];
    const float* aw3 = (const float*)d_in[13];
    const float* ab3 = (const float*)d_in[14];
    float* out = (float*)d_out;

    // workspace layout (256B aligned)
    char* ws = (char*)d_ws;
    size_t off = 0;
    auto nxt = [&](size_t bytes) {
        size_t o = off;
        off = (off + bytes + 255) & ~(size_t)255;
        return o;
    };
    size_t o_wp2 = nxt((size_t)25 * 128 * 64 * 2);
    size_t o_wp3 = nxt((size_t)25 * 256 * 128 * 2);
    size_t o_a1  = nxt((size_t)16 * 108 * 108 * 64 * 2);
    size_t o_c2  = nxt((size_t)16 * 104 * 104 * 128 * 2);
    size_t o_a2  = nxt((size_t)16 * 52 * 52 * 128 * 2);
    size_t o_c3  = nxt((size_t)16 * 48 * 48 * 256 * 2);
    size_t o_a3  = nxt((size_t)16 * 24 * 24 * 256 * 2);
    size_t o_g   = nxt((size_t)16 * 512 * 4);

    _Float16* wp2 = (_Float16*)(ws + o_wp2);
    _Float16* wp3 = (_Float16*)(ws + o_wp3);
    _Float16* a1  = (_Float16*)(ws + o_a1);
    _Float16* c2  = (_Float16*)(ws + o_c2);
    _Float16* a2  = (_Float16*)(ws + o_a2);
    _Float16* c3  = (_Float16*)(ws + o_c3);
    _Float16* a3  = (_Float16*)(ws + o_a3);
    float*    g   = (float*)(ws + o_g);

    // 0) zero g
    zero_f32_kernel<<<(16 * 512 + 255) / 256, 256, 0, stream>>>(g, 16 * 512);

    // 1) pack conv weights to f16 [tap][O][I]
    pack_w_kernel<<<(25 * 128 * 64 + 255) / 256, 256, 0, stream>>>(w2, wp2, 128, 64);
    pack_w_kernel<<<(25 * 256 * 128 + 255) / 256, 256, 0, stream>>>(w3, wp3, 256, 128);

    // 2) conv1 fused bias+relu+pool -> a1 [16][108][108][64]
    conv1_kernel<<<(16 * 108 * 108 + 255) / 256, 256, 0, stream>>>(x, w1, b1, a1);

    // 3) conv2 (WMMA) -> c2 [16][104][104][128]; pool -> a2 [16][52][52][128]
    conv_wmma_kernel<64, 128, 108, 108>
        <<<dim3(169, 1, 16), 256, 0, stream>>>(a1, wp2, b2, c2);
    pool2_kernel<<<((16 * 52 * 52 * 128) + 255) / 256, 256, 0, stream>>>(
        c2, a2, 104, 104, 128, 16 * 52 * 52 * 128);

    // 4) conv3 (WMMA) -> c3 [16][48][48][256]; pool -> a3 [16][24][24][256]
    conv_wmma_kernel<128, 256, 52, 52>
        <<<dim3(36, 2, 16), 256, 0, stream>>>(a2, wp3, b3, c3);
    pool2_kernel<<<((16 * 24 * 24 * 256) + 255) / 256, 256, 0, stream>>>(
        c3, a3, 48, 48, 256, 16 * 24 * 24 * 256);

    // 5) FC -> g [16][512]
    fc_kernel<<<dim3(512, 72), 256, 0, stream>>>(fcw, a3, g);

    // 6) attention read-outs -> out [16][448]
    att_kernel<<<16, 256, 0, stream>>>(a1,  64, 108 * 108, aw1, ab1, g, fcb, out, 0);
    att_kernel<<<16, 256, 0, stream>>>(a2, 128,  52 * 52,  aw2, ab2, g, fcb, out, 64);
    att_kernel<<<16, 256, 0, stream>>>(a3, 256,  24 * 24,  aw3, ab3, g, fcb, out, 192);
}
